// NoiseNet_6622839570536
// MI455X (gfx1250) — compile-verified, matrix operand order checked by reading, not run on hardware
//
#include <hip/hip_runtime.h>

typedef _Float16 v16h __attribute__((ext_vector_type(16)));
typedef _Float16 v8h  __attribute__((ext_vector_type(8)));
typedef float    v8f  __attribute__((ext_vector_type(8)));
typedef float    v4f  __attribute__((ext_vector_type(4)));

#define D_FEAT 64
#define EDGE_DIM 64

// tanh(a + b) with the bias add folded into the exp2 argument.
// Prefers hardware v_tanh_f32 (gfx1250) when the builtin exists; else
// tanh(y) = 1 - 2 * rcp(exp2(y * 2*log2(e)) + 1)  -> v_fma, v_exp, v_add, v_rcp, v_fma.
__device__ __forceinline__ float tanh_bias(float a, float b, float bsc) {
#if __has_builtin(__builtin_amdgcn_tanhf)
    return __builtin_amdgcn_tanhf(a + b);
#else
    float arg = __builtin_fmaf(a, 2.88539008177792681472f /* 2*log2(e) */, bsc);
#if __has_builtin(__builtin_amdgcn_exp2f)
    float e2 = __builtin_amdgcn_exp2f(arg);
#else
    float e2 = __builtin_exp2f(arg);
#endif
#if __has_builtin(__builtin_amdgcn_rcpf)
    return __builtin_fmaf(-2.0f, __builtin_amdgcn_rcpf(e2 + 1.0f), 1.0f);
#else
    return 1.0f - 2.0f / (e2 + 1.0f);
#endif
#endif
}

// One-time f32 -> f16 conversion of the node-feature table into workspace.
// 12.8 MB read + 6.4 MB write: ~1 us against the 23.3 TB/s HBM roof; removes
// 32 v_cvt_pk per tile from the hot loop and halves L2 gather traffic.
__global__ __launch_bounds__(256) void cvt_nodefeats_f16_kernel(
    const float* __restrict__ in, _Float16* __restrict__ out, int n)
{
    int stride = (int)(gridDim.x * blockDim.x) * 8;
    for (int i = (int)(blockIdx.x * blockDim.x + threadIdx.x) * 8; i < n; i += stride) {
        const v4f* p = (const v4f*)(in + i);
        v4f a = p[0], c = p[1];
        v8h h;
        h[0] = (_Float16)a.x; h[1] = (_Float16)a.y;
        h[2] = (_Float16)a.z; h[3] = (_Float16)a.w;
        h[4] = (_Float16)c.x; h[5] = (_Float16)c.y;
        h[6] = (_Float16)c.z; h[7] = (_Float16)c.w;
        *(v8h*)(out + i) = h;
    }
}

// D[16x64] = A[16x128] (gathered node feats) x W[128x64] (f16) ; tanh(+b)
// One wave per 16-edge tile; 4 N-tiles x 4 K-steps of v_wmma_f32_16x16x32_f16.
// F16G: gather pre-converted f16 features (two global_load_b128 per K-block,
// zero conversion VALU); otherwise gather f32 + convert inline.
template <bool F16G>
__global__ __launch_bounds__(256) void edge_mlp_wmma_kernel(
    const float*    __restrict__ nf32,      // [N_NODES, 64] f32
    const _Float16* __restrict__ nf16,      // [N_NODES, 64] f16 (ws) or null
    const int*      __restrict__ senders,   // [E]
    const int*      __restrict__ receivers, // [E]
    const float*    __restrict__ W,         // [128, 64] row-major
    const float*    __restrict__ bvec,      // [64]
    float*          __restrict__ out,       // [E, 64]
    int nEdges)
{
    // W staged in f16, pre-swizzled into WMMA B-fragment layout:
    // fragment f = kb*4+nb; per lane l: 16 halves = W[k][n],
    //   n = nb*16 + (l&15), k = kb*32 + (j<8 ? kloc+j : 16+kloc+(j-8)),
    //   kloc = (l>>4)*8.  (Compiler promotes fragments to registers.)
    __shared__ __align__(32) _Float16 wlds[16 * 32 * 16];  // 16 KB

    const int tid = threadIdx.x;
    for (int idx = tid; idx < 16 * 32 * 16; idx += 256) {
        int f  = idx >> 9;
        int l  = (idx >> 4) & 31;
        int j  = idx & 15;
        int kb = f >> 2, nb = f & 3;
        int kl = (l >> 4) * 8;
        int k  = kb * 32 + ((j < 8) ? (kl + j) : (16 + kl + (j - 8)));
        int n  = nb * 16 + (l & 15);
        wlds[idx] = (_Float16)W[k * EDGE_DIM + n];
    }
    __syncthreads();

    const int lane  = tid & 31;
    const int m     = lane & 15;
    const int khalf = lane >> 4;
    const int kloc  = khalf * 8;

    float bvals[4], bsc[4];
    #pragma unroll
    for (int nb = 0; nb < 4; ++nb) {
        bvals[nb] = bvec[nb * 16 + m];
        bsc[nb]   = bvals[nb] * 2.88539008177792681472f;  // b * 2*log2(e)
    }

    const v16h* bfrags = (const v16h*)wlds;

    const int nTiles = (nEdges + 15) >> 4;
    const int waveId = (int)((blockIdx.x * blockDim.x + tid) >> 5);
    const int nWaves = (int)((gridDim.x * blockDim.x) >> 5);

    for (int tile = waveId; tile < nTiles; tile += nWaves) {
        const int e0 = tile << 4;
        int e = e0 + m;
        if (e >= nEdges) e = nEdges - 1;  // tail clamp (E%16==0 in practice)
        const int rn = receivers[e];
        const int sn = senders[e];

        // A fragments: K in [0,64) = recv feats, [64,128) = send feats.
        v16h A[4];
        #pragma unroll
        for (int kb = 0; kb < 4; ++kb) {
            const int node = (kb < 2) ? rn : sn;
            if (F16G) {
                const v8h* p = (const v8h*)(nf16 + (size_t)node * D_FEAT
                                            + (kb & 1) * 32 + kloc);
                union { v16h v; v8h h[2]; } u;
                u.h[0] = p[0];   // cols kloc .. kloc+7
                u.h[1] = p[2];   // cols kloc+16 .. kloc+23
                A[kb] = u.v;
            } else {
                const v4f* p = (const v4f*)(nf32 + (size_t)node * D_FEAT
                                            + (kb & 1) * 32 + kloc);
                v4f f0 = p[0], f1 = p[1], f2 = p[4], f3 = p[5];
                v16h a;
                a[0]  = (_Float16)f0.x; a[1]  = (_Float16)f0.y;
                a[2]  = (_Float16)f0.z; a[3]  = (_Float16)f0.w;
                a[4]  = (_Float16)f1.x; a[5]  = (_Float16)f1.y;
                a[6]  = (_Float16)f1.z; a[7]  = (_Float16)f1.w;
                a[8]  = (_Float16)f2.x; a[9]  = (_Float16)f2.y;
                a[10] = (_Float16)f2.z; a[11] = (_Float16)f2.w;
                a[12] = (_Float16)f3.x; a[13] = (_Float16)f3.y;
                a[14] = (_Float16)f3.z; a[15] = (_Float16)f3.w;
                A[kb] = a;
            }
        }

        v8f acc[4] = {};
        #pragma unroll
        for (int nb = 0; nb < 4; ++nb) {
            #pragma unroll
            for (int kb = 0; kb < 4; ++kb) {
                v16h bf = bfrags[(kb * 4 + nb) * 32 + lane];
                acc[nb] = __builtin_amdgcn_wmma_f32_16x16x32_f16(
                    false, A[kb], false, bf, (short)0, acc[nb], false, false);
            }
        }

        // C/D layout: VGPR r holds row (r + 8*khalf), col = nb*16 + m.
        float* obase = out + (size_t)(e0 + 8 * khalf) * EDGE_DIM + m;
        if (e0 + 16 <= nEdges) {           // wave-uniform fast path, no guards
            #pragma unroll
            for (int nb = 0; nb < 4; ++nb) {
                #pragma unroll
                for (int r = 0; r < 8; ++r) {
                    obase[(size_t)r * EDGE_DIM + nb * 16] =
                        tanh_bias(acc[nb][r], bvals[nb], bsc[nb]);
                }
            }
        } else {                            // tail tile
            #pragma unroll
            for (int nb = 0; nb < 4; ++nb) {
                #pragma unroll
                for (int r = 0; r < 8; ++r) {
                    if (e0 + r + 8 * khalf < nEdges) {
                        obase[(size_t)r * EDGE_DIM + nb * 16] =
                            tanh_bias(acc[nb][r], bvals[nb], bsc[nb]);
                    }
                }
            }
        }
    }
}

extern "C" void kernel_launch(void* const* d_in, const int* in_sizes, int n_in,
                              void* d_out, int out_size, void* d_ws, size_t ws_size,
                              hipStream_t stream) {
    const float* node_feats = (const float*)d_in[0];
    const int*   senders    = (const int*)d_in[1];
    const int*   receivers  = (const int*)d_in[2];
    const float* W          = (const float*)d_in[3];
    const float* b          = (const float*)d_in[4];
    float*       out        = (float*)d_out;

    const int nFeatElems = in_sizes[0];     // N_NODES * 64
    const int nEdges     = in_sizes[1];

    const int nTiles = (nEdges + 15) / 16;
    int blocks = (nTiles + 7) / 8;          // 8 waves per 256-thread block
    if (blocks > 2048) blocks = 2048;
    if (blocks < 1) blocks = 1;

    const size_t needF16 = (size_t)nFeatElems * sizeof(unsigned short);
    if (ws_size >= needF16) {
        _Float16* nf16 = (_Float16*)d_ws;
        int cblocks = (nFeatElems / 8 + 255) / 256;
        if (cblocks > 4096) cblocks = 4096;
        if (cblocks < 1) cblocks = 1;
        cvt_nodefeats_f16_kernel<<<cblocks, 256, 0, stream>>>(node_feats, nf16, nFeatElems);
        edge_mlp_wmma_kernel<true><<<blocks, 256, 0, stream>>>(
            node_feats, nf16, senders, receivers, W, b, out, nEdges);
    } else {
        edge_mlp_wmma_kernel<false><<<blocks, 256, 0, stream>>>(
            node_feats, (const _Float16*)nullptr, senders, receivers, W, b, out, nEdges);
    }
}
